// BinarizedLinear_81999515615872
// MI455X (gfx1250) — matrix operand order repaired and checked by reading, not compiled, not run on hardware
//
#include <hip/hip_runtime.h>
#include <stdint.h>
#include <stddef.h>

// ---- types ----------------------------------------------------------------
typedef __attribute__((ext_vector_type(16))) __bf16        v16bf;
typedef __attribute__((ext_vector_type(8)))  float         v8f;
typedef __attribute__((ext_vector_type(4)))  unsigned int  u32x4;
typedef __attribute__((ext_vector_type(4)))  float         f32x4;

struct Pack32 { u32x4 lo; u32x4 hi; };   // 32 bytes -> bit_cast to v16bf

// ---- problem constants ----------------------------------------------------
#define IN_FEAT   1024
#define OUT_FEAT  1024
#define M_TILE    128
#define N_TILE    128
#define K_TILE    32
#define LDS_K     40           // K_TILE + 8 bf16 pad -> 80B row pitch, conflict-free b128 reads

__device__ __forceinline__ unsigned short cvt_bf16_rne(float f) {
  uint32_t u = __float_as_uint(f);
  u += 0x7FFFu + ((u >> 16) & 1u);       // round-to-nearest-even
  return (unsigned short)(u >> 16);
}

__device__ __forceinline__ unsigned short binarize_bf16(float w) {
  // sign(w) with sign(+-0) -> +1 ; bf16 bit patterns for +-1.0
  return (w < 0.0f) ? (unsigned short)0xBF80u : (unsigned short)0x3F80u;
}

__global__ __launch_bounds__(256)
void binlin_wmma_bf16(const float* __restrict__ X,
                      const float* __restrict__ W,
                      const float* __restrict__ Bias,
                      float* __restrict__ Out) {
  // double-buffered LDS tiles (bf16 bits), 40 KB total
  __shared__ unsigned short As[2][M_TILE * LDS_K];   // [buf][row (batch)][k]
  __shared__ unsigned short Bs[2][N_TILE * LDS_K];   // [buf][out-feature][k]

  const int tid   = threadIdx.x;
  const int lane  = tid & 31;
  const int wave  = tid >> 5;        // 0..7
  const int waveM = wave & 3;        // 4 waves along M (32 rows each)
  const int waveN = wave >> 2;       // 2 waves along N (64 cols each)

  const int blockN = blockIdx.x * N_TILE;   // fastest dim: same-M blocks adjacent -> x reuse in L2
  const int blockM = blockIdx.y * M_TILE;

  const int r16   = lane & 15;
  const int khalf = lane >> 4;       // 0 or 1

  v8f acc[2][4];
  #pragma unroll
  for (int i = 0; i < 2; ++i)
    #pragma unroll
    for (int j = 0; j < 4; ++j)
      #pragma unroll
      for (int p = 0; p < 8; ++p)
        acc[i][j][p] = 0.0f;

  // global -> register staging map: 128 rows x 32 f32 per tile, 256 threads,
  // each thread carries 4 float4's per matrix (one per 32-row stripe).
  const int col4 = tid & 7;          // which float4 inside the 32-float row
  const int row0 = tid >> 3;         // 0..31 ; rows row0 + {0,32,64,96}

  f32x4 xa[4];                       // staged x tile
  f32x4 wb[4];                       // staged weight tile

  // ---- prologue: stage K-tile 0 ----
  #pragma unroll
  for (int r = 0; r < 4; ++r) {
    const int row = row0 + 32 * r;
    xa[r] = *reinterpret_cast<const f32x4*>(
        X + (size_t)(blockM + row) * IN_FEAT + col4 * 4);
    wb[r] = *reinterpret_cast<const f32x4*>(
        W + (size_t)(blockN + row) * IN_FEAT + col4 * 4);
  }
  // convert + binarize into buffer 0
  #pragma unroll
  for (int r = 0; r < 4; ++r) {
    const int row = row0 + 32 * r;
    unsigned short* da = &As[0][row * LDS_K + col4 * 4];
    da[0] = cvt_bf16_rne(xa[r].x);  da[1] = cvt_bf16_rne(xa[r].y);
    da[2] = cvt_bf16_rne(xa[r].z);  da[3] = cvt_bf16_rne(xa[r].w);
    unsigned short* db = &Bs[0][row * LDS_K + col4 * 4];
    db[0] = binarize_bf16(wb[r].x); db[1] = binarize_bf16(wb[r].y);
    db[2] = binarize_bf16(wb[r].z); db[3] = binarize_bf16(wb[r].w);
  }

  int buf = 0;
  for (int k0 = 0; k0 < IN_FEAT; k0 += K_TILE) {
    __syncthreads();   // writes to `buf` visible; reads of `buf^1` (prev step) done

    const bool more = (k0 + K_TILE) < IN_FEAT;

    // ---- issue next tile's global loads NOW (overlap with WMMA below) ----
    if (more) {
      #pragma unroll
      for (int r = 0; r < 4; ++r) {
        const int row = row0 + 32 * r;
        xa[r] = *reinterpret_cast<const f32x4*>(
            X + (size_t)(blockM + row) * IN_FEAT + (k0 + K_TILE) + col4 * 4);
        wb[r] = *reinterpret_cast<const f32x4*>(
            W + (size_t)(blockN + row) * IN_FEAT + (k0 + K_TILE) + col4 * 4);
      }
    }

    // ---- A fragments: 16x32 bf16, ISA wave32 layout ----
    // lane L: row = L%16 ; elems 0..7 <- K = (L/16)*8 ; elems 8..15 <- K = 16+(L/16)*8
    v16bf afrag[2];
    #pragma unroll
    for (int i = 0; i < 2; ++i) {
      const unsigned short* base =
          &As[buf][(waveM * 32 + i * 16 + r16) * LDS_K + khalf * 8];
      Pack32 p;
      p.lo = *reinterpret_cast<const u32x4*>(base);        // K = khalf*8 + 0..7
      p.hi = *reinterpret_cast<const u32x4*>(base + 16);   // K = 16 + khalf*8 + 0..7
      afrag[i] = __builtin_bit_cast(v16bf, p);
    }
    // ---- B fragments: 32x16 bf16 ----
    // lane L: col(out-feat) = L%16 ; elems 0..7 <- K = (L/16)*8 ; elems 8..15 <- K = 16+(L/16)*8
    v16bf bfrag[4];
    #pragma unroll
    for (int j = 0; j < 4; ++j) {
      const unsigned short* base =
          &Bs[buf][(waveN * 64 + j * 16 + r16) * LDS_K + khalf * 8];
      Pack32 p;
      p.lo = *reinterpret_cast<const u32x4*>(base);        // K = khalf*8 + 0..7
      p.hi = *reinterpret_cast<const u32x4*>(base + 16);   // K = 16 + khalf*8 + 0..7
      bfrag[j] = __builtin_bit_cast(v16bf, p);
    }

    // ---- 8 WMMAs per K-step: D = A x B + C, f32 accumulate ----
    #pragma unroll
    for (int i = 0; i < 2; ++i)
      #pragma unroll
      for (int j = 0; j < 4; ++j)
        acc[i][j] = __builtin_amdgcn_wmma_f32_16x16x32_bf16(
            false, afrag[i], false, bfrag[j], (short)0, acc[i][j], false, false);

    // ---- convert staged registers into the other buffer ----
    if (more) {
      const int nb = buf ^ 1;
      #pragma unroll
      for (int r = 0; r < 4; ++r) {
        const int row = row0 + 32 * r;
        unsigned short* da = &As[nb][row * LDS_K + col4 * 4];
        da[0] = cvt_bf16_rne(xa[r].x);  da[1] = cvt_bf16_rne(xa[r].y);
        da[2] = cvt_bf16_rne(xa[r].z);  da[3] = cvt_bf16_rne(xa[r].w);
        unsigned short* db = &Bs[nb][row * LDS_K + col4 * 4];
        db[0] = binarize_bf16(wb[r].x); db[1] = binarize_bf16(wb[r].y);
        db[2] = binarize_bf16(wb[r].z); db[3] = binarize_bf16(wb[r].w);
      }
    }
    buf ^= 1;
  }

  // ---- epilogue: C/D layout: VGPR p -> M = p + 8*khalf, N = lane%16 ----
  #pragma unroll
  for (int j = 0; j < 4; ++j) {
    const int col = blockN + waveN * 64 + j * 16 + r16;
    const float b = Bias[col];
    #pragma unroll
    for (int i = 0; i < 2; ++i) {
      const int rowBase = blockM + waveM * 32 + i * 16 + khalf * 8;
      #pragma unroll
      for (int p = 0; p < 8; ++p) {
        __builtin_nontemporal_store(acc[i][j][p] + b,
            Out + (size_t)(rowBase + p) * OUT_FEAT + col);
      }
    }
  }
}

// ---- host-side launcher ---------------------------------------------------
extern "C" void kernel_launch(void* const* d_in, const int* in_sizes, int n_in,
                              void* d_out, int out_size, void* d_ws, size_t ws_size,
                              hipStream_t stream) {
  const float* X    = (const float*)d_in[0];   // (BATCH, 1024) f32
  const float* W    = (const float*)d_in[1];   // (1024, 1024)  f32
  const float* Bias = (const float*)d_in[2];   // (1024,)       f32
  float* Out        = (float*)d_out;           // (BATCH, 1024) f32

  const int batch = in_sizes[0] / IN_FEAT;     // 32768
  dim3 grid(OUT_FEAT / N_TILE, batch / M_TILE);  // (8, 256): N fastest for x L2 reuse
  dim3 block(256);                               // 8 wave32's
  hipLaunchKernelGGL(binlin_wmma_bf16, grid, block, 0, stream, X, W, Bias, Out);
}
